// NameFormatModel_79920751444272
// MI455X (gfx1250) — matrix operand order repaired and checked by reading, not compiled, not run on hardware
//
#include <hip/hip_runtime.h>
#include <hip/hip_bf16.h>
#include <math.h>

// ---------------------------------------------------------------------------
// MI455X (gfx1250) bidirectional-LSTM inference.
//  * bf16 WMMA 16x16x32 (f32 accum) for all gate GEMMs
//  * per-wave ownership of matching i/f/g/o columns -> in-register cell update
//  * 128-thread blocks (4 wave32) x 128 WGs per step for occupancy
//  * async global->LDS staging of A-tiles (ASYNCcnt) when toolchain exposes it
// ---------------------------------------------------------------------------

typedef __bf16 bf16;
typedef __attribute__((ext_vector_type(16))) __bf16 v16bf;
typedef __attribute__((ext_vector_type(8)))  float  v8f;

#define SEQ_LEN   128
#define BATCH     256
#define HID       512
#define NLAYERS   4
#define EMB       16
#define OUT_SZ    64
#define G4H       (4*HID)          // 2048 gate width
#define BLK       128              // 4 waves per workgroup
#define CSPLIT    4                // column-splits per (rowgroup, dir)

// ---- async global->LDS staging (gfx1250 ASYNCcnt path), with fallback -----
#if defined(__has_builtin)
#if __has_builtin(__builtin_amdgcn_global_load_async_to_lds_b128)
#define HAVE_ASYNC_LDS 1
#endif
#endif
#ifndef HAVE_ASYNC_LDS
#define HAVE_ASYNC_LDS 0
#endif

#if HAVE_ASYNC_LDS
// builtin expects pointers to the 128-bit payload type (v4i32), global + LDS AS
typedef int v4i32 __attribute__((vector_size(16)));
typedef __attribute__((address_space(1))) v4i32 ga_v4i;
typedef __attribute__((address_space(3))) v4i32 ls_v4i;
__device__ __forceinline__ void async_copy16(void* l, const void* g) {
    __builtin_amdgcn_global_load_async_to_lds_b128((ga_v4i*)g, (ls_v4i*)l, 0, 0);
}
__device__ __forceinline__ void async_wait0() {
#if __has_builtin(__builtin_amdgcn_s_wait_asynccnt)
    __builtin_amdgcn_s_wait_asynccnt(0);
#else
    asm volatile("s_wait_asynccnt 0" ::: "memory");
#endif
}
#endif

__device__ __forceinline__ float sigmoidf_(float x) {
    return 1.0f / (1.0f + __expf(-x));
}

// A-matrix fragment (16x32 bf16), ISA layout:
//  lanes 0-15  : row M=lane,    K = [0..7]  then [16..23]
//  lanes 16-31 : row M=lane-16, K = [8..15] then [24..31]
__device__ __forceinline__ v16bf load_a_frag(const bf16* rowbase, int k0, int hi) {
    v16bf a;
    __builtin_memcpy(&a,             rowbase + k0 +      hi * 8, 16);
    __builtin_memcpy((char*)&a + 16, rowbase + k0 + 16 + hi * 8, 16);
    return a;
}

// B-matrix fragment (32x16 bf16): lanes 0-15 hold K=[0..7] then [16..23] of column N=lane,
// lanes 16-31 hold K=[8..15] then [24..31]  => two 16B reads from weight row N.
__device__ __forceinline__ v16bf load_b_frag(const bf16* p, int hi) {
    v16bf b;
    __builtin_memcpy(&b,             p +      hi * 8, 16);
    __builtin_memcpy((char*)&b + 16, p + 16 + hi * 8, 16);
    return b;
}

// ---------------------------------------------------------------------------
// fp32 -> bf16 weight conversion with optional K padding (layer0: 16 -> 32)
// ---------------------------------------------------------------------------
__global__ void cvt_pad_bf16(const float* __restrict__ src, bf16* __restrict__ dst,
                             long rows, int K, int Kp) {
    long total = rows * (long)Kp;
    for (long i = blockIdx.x * (long)blockDim.x + threadIdx.x; i < total;
         i += (long)gridDim.x * blockDim.x) {
        long row = i / Kp;
        int  col = (int)(i - row * Kp);
        dst[i] = (col < K) ? (bf16)src[row * (long)K + col] : (bf16)0.0f;
    }
}

// combined bias  b = bih + bhh  ->  [L][2][4H] fp32
__global__ void bias_combine(const float* __restrict__ bih0, const float* __restrict__ bhh0,
                             const float* __restrict__ bih,  const float* __restrict__ bhh,
                             float* __restrict__ out) {
    int i = blockIdx.x * blockDim.x + threadIdx.x;
    if (i >= NLAYERS * 2 * G4H) return;
    int l = i / (2 * G4H);
    int rem = i - l * (2 * G4H);
    out[i] = (l == 0) ? (bih0[rem] + bhh0[rem])
                      : (bih[(l - 1) * 2 * G4H + rem] + bhh[(l - 1) * 2 * G4H + rem]);
}

// embedding lookup -> bf16 xs [T][B][EMB]
__global__ void embed_lookup(const int* __restrict__ tokens, const float* __restrict__ emb,
                             bf16* __restrict__ xs, int total) {
    int i = blockIdx.x * blockDim.x + threadIdx.x;
    if (i >= total) return;
    int tok = tokens[i];
    #pragma unroll
    for (int e = 0; e < EMB; ++e)
        xs[(size_t)i * EMB + e] = (bf16)emb[tok * EMB + e];
}

// ---------------------------------------------------------------------------
// One (timestep, layer) for both directions.
//  grid = (16 row-groups of 16 batch rows, 2 dirs * 4 colsplits), block = 128
//  wave w of colsplit cs owns columns [32*(4cs+w), +32) of each gate block
// ---------------------------------------------------------------------------
__global__ __launch_bounds__(BLK)
void lstm_step(const bf16* __restrict__ x, int Kx, int xStride,
               const bf16* __restrict__ Wi, int WiStride,   // padded K (>=32)
               const bf16* __restrict__ Wh,
               const float* __restrict__ bias,              // [2][4H]
               bf16*  __restrict__ hB,                      // [B][2][HID] bf16 state
               float* __restrict__ hF,                      // [2][B][HID] fp32 copy
               float* __restrict__ cS)                      // [2][B][HID]
{
    extern __shared__ char smem[];
    const int d    = blockIdx.y >> 2;          // direction
    const int cs   = blockIdx.y & 3;           // column split
    const int r    = blockIdx.x;               // row group
    const int tid  = threadIdx.x;
    const int lane = tid & 31;
    const int wave = tid >> 5;                 // 0..3
    const int Kxp  = WiStride;

    bf16* xlds = (bf16*)smem;                  // [16][Kxp]
    bf16* hlds = xlds + 16 * Kxp;              // [16][HID]

    const bf16*  Wi_d = Wi + (size_t)d * G4H * WiStride;
    const bf16*  Wh_d = Wh + (size_t)d * G4H * HID;
    const float* b_d  = bias + d * G4H;
    float* c_d  = cS + (size_t)d * BATCH * HID;
    float* hF_d = hF + (size_t)d * BATCH * HID;

    // ---- stage x tile [16 x Kxp] (zero-padded) and h tile into LDS --------
#if HAVE_ASYNC_LDS
    if (Kx == Kxp) {
        // async path: 16B chunks, per-lane LDS destinations, ASYNCcnt tracked
        const int perRowX = Kx >> 3;                       // 16B chunks per row
        for (int i = tid; i < 16 * perRowX; i += BLK) {
            int row = i / perRowX, c = i - row * perRowX;
            async_copy16(xlds + (size_t)row * Kxp + c * 8,
                         x + (size_t)(16 * r + row) * xStride + c * 8);
        }
        for (int i = tid; i < 16 * 64; i += BLK) {         // h: 64 chunks/row
            int row = i >> 6, c = i & 63;
            async_copy16(hlds + (size_t)row * HID + c * 8,
                         hB + (((size_t)(16 * r + row)) * 2 + d) * HID + c * 8);
        }
        async_wait0();
    } else
#endif
    {
        for (int i = tid; i < 16 * Kxp; i += BLK) {
            int row = i / Kxp, col = i - row * Kxp;
            xlds[i] = (col < Kx) ? x[(size_t)(16 * r + row) * xStride + col] : (bf16)0.0f;
        }
        for (int i = tid; i < 16 * HID; i += BLK) {
            int row = i >> 9, col = i & (HID - 1);
            hlds[i] = hB[(((size_t)(16 * r + row)) * 2 + d) * HID + col];
        }
    }
    __syncthreads();

    const int cb = (cs * 4 + wave) * 32;       // column base within gate block
    const int hi = lane >> 4;
    const int ln = lane & 15;

    // accumulators seeded with the (per-column) combined bias
    v8f acc[4][2];
    #pragma unroll
    for (int g = 0; g < 4; ++g)
        #pragma unroll
        for (int q = 0; q < 2; ++q) {
            float bv = b_d[g * HID + cb + q * 16 + ln];
            #pragma unroll
            for (int e = 0; e < 8; ++e) acc[g][q][e] = bv;
        }

    // ---- gates += x @ Wi^T  (K = Kxp) ----
    const bf16* arow_x = xlds + (size_t)ln * Kxp;
    for (int k0 = 0; k0 < Kxp; k0 += 32) {
        v16bf a = load_a_frag(arow_x, k0, hi);
        __builtin_prefetch(Wi_d + (size_t)(cb + ln) * WiStride + k0 + 64, 0, 1);
        #pragma unroll
        for (int g = 0; g < 4; ++g) {
            #pragma unroll
            for (int q = 0; q < 2; ++q) {
                int N = g * HID + cb + q * 16 + ln;
                v16bf b = load_b_frag(Wi_d + (size_t)N * WiStride + k0, hi);
                acc[g][q] = __builtin_amdgcn_wmma_f32_16x16x32_bf16(
                    false, a, false, b, (short)0, acc[g][q], false, false);
            }
        }
    }

    // ---- gates += h @ Wh^T  (K = 512) ----
    const bf16* arow_h = hlds + (size_t)ln * HID;
    for (int k0 = 0; k0 < HID; k0 += 32) {
        v16bf a = load_a_frag(arow_h, k0, hi);
        __builtin_prefetch(Wh_d + (size_t)(cb + ln) * HID + k0 + 64, 0, 1);
        #pragma unroll
        for (int g = 0; g < 4; ++g) {
            #pragma unroll
            for (int q = 0; q < 2; ++q) {
                int N = g * HID + cb + q * 16 + ln;
                v16bf b = load_b_frag(Wh_d + (size_t)N * HID + k0, hi);
                acc[g][q] = __builtin_amdgcn_wmma_f32_16x16x32_bf16(
                    false, a, false, b, (short)0, acc[g][q], false, false);
            }
        }
    }

    // ---- in-register LSTM cell update (wave owns matching i/f/g/o cols) ----
    // D layout: vgpr j, lanes 0-15 -> (M=j, N=ln); lanes 16-31 -> (M=j+8, N=ln)
    #pragma unroll
    for (int q = 0; q < 2; ++q) {
        int n = cb + q * 16 + ln;
        #pragma unroll
        for (int j = 0; j < 8; ++j) {
            int m = j + hi * 8;
            int brow = 16 * r + m;
            float iv = sigmoidf_(acc[0][q][j]);
            float fv = sigmoidf_(acc[1][q][j]);
            float gv = tanhf(acc[2][q][j]);
            float ov = sigmoidf_(acc[3][q][j]);
            size_t ci = (size_t)brow * HID + n;
            float cn = fv * c_d[ci] + iv * gv;
            float hn = ov * tanhf(cn);
            c_d[ci]  = cn;
            hF_d[ci] = hn;
            hB[((size_t)brow * 2 + d) * HID + n] = (bf16)hn;
        }
    }
}

// ---------------------------------------------------------------------------
// final FC (4096 -> 64) + softmax.  hid row b = concat over s=2l+d of [h|c].
// ---------------------------------------------------------------------------
__global__ __launch_bounds__(OUT_SZ)
void fc_softmax(const float* __restrict__ hF, const float* __restrict__ cS,
                const float* __restrict__ fcw, const float* __restrict__ fcb,
                float* __restrict__ out) {
    int b = blockIdx.x, o = threadIdx.x;
    float acc = fcb[o];
    for (int s = 0; s < 2 * NLAYERS; ++s) {
        const float* hrow = hF + ((size_t)s * BATCH + b) * HID;
        const float* crow = cS + ((size_t)s * BATCH + b) * HID;
        const float* w    = fcw + (size_t)o * 4096 + s * 2 * HID;
        for (int k = 0; k < HID; ++k)
            acc += hrow[k] * w[k] + crow[k] * w[HID + k];
    }
    __shared__ float red[OUT_SZ];
    red[o] = acc;
    __syncthreads();
    for (int off = 32; off > 0; off >>= 1) {
        if (o < off) red[o] = fmaxf(red[o], red[o + off]);
        __syncthreads();
    }
    float mx = red[0];
    __syncthreads();
    float e = __expf(acc - mx);
    red[o] = e;
    __syncthreads();
    for (int off = 32; off > 0; off >>= 1) {
        if (o < off) red[o] += red[o + off];
        __syncthreads();
    }
    out[(size_t)b * OUT_SZ + o] = e / red[0];
}

// ---------------------------------------------------------------------------
extern "C" void kernel_launch(void* const* d_in, const int* in_sizes, int n_in,
                              void* d_out, int out_size, void* d_ws, size_t ws_size,
                              hipStream_t stream) {
    (void)in_sizes; (void)n_in; (void)out_size; (void)ws_size;
    const int*   tokens = (const int*)  d_in[0];
    const float* emb    = (const float*)d_in[1];
    const float* Wih0   = (const float*)d_in[2];
    const float* Whh0   = (const float*)d_in[3];
    const float* bih0   = (const float*)d_in[4];
    const float* bhh0   = (const float*)d_in[5];
    const float* Wih    = (const float*)d_in[6];
    const float* Whh    = (const float*)d_in[7];
    const float* bih    = (const float*)d_in[8];
    const float* bhh    = (const float*)d_in[9];
    const float* fcw    = (const float*)d_in[10];
    const float* fcb    = (const float*)d_in[11];
    float* out = (float*)d_out;

    // ---- workspace layout ----
    char* ws = (char*)d_ws;
    size_t off = 0;
    auto alloc = [&](size_t bytes) -> void* {
        off = (off + 255) & ~(size_t)255;
        void* p = ws + off;
        off += bytes;
        return p;
    };
    bf16*  wih0_b = (bf16*) alloc((size_t)2 * G4H * 32   * sizeof(bf16));  // K padded 16->32
    bf16*  whh0_b = (bf16*) alloc((size_t)2 * G4H * HID  * sizeof(bf16));
    bf16*  wih_b  = (bf16*) alloc((size_t)3 * 2 * G4H * 2 * HID * sizeof(bf16));
    bf16*  whh_b  = (bf16*) alloc((size_t)3 * 2 * G4H * HID * sizeof(bf16));
    float* bias   = (float*)alloc((size_t)NLAYERS * 2 * G4H * sizeof(float));
    bf16*  xs     = (bf16*) alloc((size_t)SEQ_LEN * BATCH * EMB * sizeof(bf16));
    bf16*  hB     = (bf16*) alloc((size_t)NLAYERS * BATCH * 2 * HID * sizeof(bf16));
    float* hF     = (float*)alloc((size_t)NLAYERS * 2 * BATCH * HID * sizeof(float));
    float* cS     = (float*)alloc((size_t)NLAYERS * 2 * BATCH * HID * sizeof(float));

    // ---- weight conversion (bf16, layer0 K padded to 32) ----
    {
        long n;
        n = (long)2 * G4H * 32;
        cvt_pad_bf16<<<(unsigned)((n + 255) / 256), 256, 0, stream>>>(Wih0, wih0_b, 2 * G4H, EMB, 32);
        n = (long)2 * G4H * HID;
        cvt_pad_bf16<<<(unsigned)((n + 255) / 256), 256, 0, stream>>>(Whh0, whh0_b, 2 * G4H, HID, HID);
        n = (long)3 * 2 * G4H * 2 * HID;
        cvt_pad_bf16<<<(unsigned)((n + 255) / 256), 256, 0, stream>>>(Wih, wih_b, (long)3 * 2 * G4H, 2 * HID, 2 * HID);
        n = (long)3 * 2 * G4H * HID;
        cvt_pad_bf16<<<(unsigned)((n + 255) / 256), 256, 0, stream>>>(Whh, whh_b, (long)3 * 2 * G4H, HID, HID);
    }
    bias_combine<<<(NLAYERS * 2 * G4H + 255) / 256, 256, 0, stream>>>(bih0, bhh0, bih, bhh, bias);
    embed_lookup<<<(SEQ_LEN * BATCH + 255) / 256, 256, 0, stream>>>(tokens, emb, xs, SEQ_LEN * BATCH);

    // ---- zero initial state ----
    (void)hipMemsetAsync(hB, 0, (size_t)NLAYERS * BATCH * 2 * HID * sizeof(bf16), stream);
    (void)hipMemsetAsync(cS, 0, (size_t)NLAYERS * 2 * BATCH * HID * sizeof(float), stream);

    // ---- recurrent loop: T x L layer-step launches (graph-captured) ----
    const size_t smem0 = (size_t)16 * (32 + HID) * sizeof(bf16);        // 17.4 KB
    const size_t smemL = (size_t)16 * (2 * HID + HID) * sizeof(bf16);   // 48 KB
    const dim3 grid(16, 2 * CSPLIT);
    for (int t = 0; t < SEQ_LEN; ++t) {
        // layer 0: x = embedded tokens (K=16 padded to 32)
        lstm_step<<<grid, BLK, smem0, stream>>>(
            xs + (size_t)t * BATCH * EMB, EMB, EMB,
            wih0_b, 32, whh0_b, bias,
            hB, hF, cS);
        for (int l = 1; l < NLAYERS; ++l) {
            lstm_step<<<grid, BLK, smemL, stream>>>(
                hB + (size_t)(l - 1) * BATCH * 2 * HID, 2 * HID, 2 * HID,
                wih_b + (size_t)(l - 1) * 2 * G4H * 2 * HID, 2 * HID,
                whh_b + (size_t)(l - 1) * 2 * G4H * HID,
                bias + (size_t)l * 2 * G4H,
                hB + (size_t)l * BATCH * 2 * HID,
                hF + (size_t)l * 2 * BATCH * HID,
                cS + (size_t)l * 2 * BATCH * HID);
        }
    }

    // ---- FC + softmax ----
    fc_softmax<<<BATCH, OUT_SZ, 0, stream>>>(hF, cS, fcw, fcb, out);
}
